// multitask_MLP_55078660603958
// MI455X (gfx1250) — matrix operand order repaired and checked by reading, not compiled, not run on hardware
//
#include <hip/hip_runtime.h>
#include <cmath>

typedef __attribute__((ext_vector_type(16))) _Float16 v16h;
typedef __attribute__((ext_vector_type(4)))  _Float16 v4h;
typedef __attribute__((ext_vector_type(8)))  float    v8f;

#define LDP 48            // LDS pitch in halves: 96B rows -> 16B-aligned b128 async writes
#define ABUF (128 * LDP)  // halves per A LDS buffer
#define BBUF (128 * LDP)  // halves per B LDS buffer

union HFrag { v16h v; v4h q[4]; };

__device__ __forceinline__ unsigned lds_addr(const void* p) {
  return (unsigned)(uintptr_t)p;   // low 32 bits of generic shared pointer = LDS byte offset
}
__device__ __forceinline__ void async_b128(unsigned lds, const void* gaddr) {
  asm volatile("global_load_async_to_lds_b128 %0, %1, off" :: "v"(lds), "v"(gaddr) : "memory");
}

// ---------------- fp32 -> f16 convert (for x) ---------------------------------------
__global__ __launch_bounds__(256)
void convert_f16(const float* __restrict__ src, _Float16* __restrict__ dst) {
  const long i = ((long)blockIdx.x * 256 + threadIdx.x) * 8;
  const float4* p = (const float4*)(src + i);
  float4 a = p[0], b = p[1];
  v4h h0 = {(_Float16)a.x, (_Float16)a.y, (_Float16)a.z, (_Float16)a.w};
  v4h h1 = {(_Float16)b.x, (_Float16)b.y, (_Float16)b.z, (_Float16)b.w};
  *(v4h*)(dst + i) = h0; *(v4h*)(dst + i + 4) = h1;
}

// ---------------- weight convert + transpose: fp32 [K][N] -> f16 [N][K] -------------
__global__ __launch_bounds__(256)
void transpose_sq_f16(const float* __restrict__ ws1, const float* __restrict__ wt1,
                      const float* __restrict__ ws2, const float* __restrict__ wt2,
                      _Float16* __restrict__ dst)
{
  const size_t MM = 1024u * 1024u;
  const int z = blockIdx.z;
  const float* src;
  if      (z < 2)  src = ws1 + (size_t)z * MM;
  else if (z < 10) src = wt1 + (size_t)(z - 2) * MM;
  else if (z < 12) src = ws2 + (size_t)(z - 10) * MM;
  else             src = wt2 + (size_t)(z - 12) * MM;
  _Float16* d = dst + (size_t)z * MM;

  __shared__ float tile[32][33];
  const int n0 = blockIdx.x * 32, k0 = blockIdx.y * 32;
  const int tx = threadIdx.x, ty = threadIdx.y;   // 32 x 8
#pragma unroll
  for (int i = 0; i < 4; ++i)
    tile[ty + 8 * i][tx] = src[(size_t)(k0 + ty + 8 * i) * 1024 + n0 + tx];
  __syncthreads();
#pragma unroll
  for (int i = 0; i < 4; ++i)
    d[(size_t)(n0 + ty + 8 * i) * 1024 + k0 + tx] = (_Float16)tile[tx][ty + 8 * i];
}

__global__ __launch_bounds__(256)
void transpose_h1_f16(const float* __restrict__ h1, _Float16* __restrict__ dst)
{
  const size_t MT = 1024u * 64u;
  const int t = blockIdx.z;
  const float* src = h1 + (size_t)t * MT;   // [1024][64]
  _Float16* d = dst + (size_t)t * MT;       // [64][1024]

  __shared__ float tile[32][33];
  const int n0 = blockIdx.x * 32, k0 = blockIdx.y * 32;
  const int tx = threadIdx.x, ty = threadIdx.y;
#pragma unroll
  for (int i = 0; i < 4; ++i)
    tile[ty + 8 * i][tx] = src[(size_t)(k0 + ty + 8 * i) * 64 + n0 + tx];
  __syncthreads();
#pragma unroll
  for (int i = 0; i < 4; ++i)
    d[(size_t)(n0 + ty + 8 * i) * 1024 + k0 + tx] = (_Float16)tile[tx][ty + 8 * i];
}

// ---------------- task routing: compact per-task row lists --------------------------
__global__ void build_lists(const int* __restrict__ bt, int B,
                            int* cnt, int* offs, int* list)
{
  if (threadIdx.x < 4) cnt[threadIdx.x] = 0;
  __syncthreads();
  for (int r = threadIdx.x; r < B; r += blockDim.x) {
    const int t = bt[r];
    const int pos = atomicAdd(&cnt[t], 1);
    list[(size_t)t * B + pos] = r;
  }
  __syncthreads();
  if (threadIdx.x == 0) {
    offs[0] = 0;
    for (int t = 0; t < 4; ++t) offs[t + 1] = offs[t] + cnt[t];
  }
}

// ---------------- main WMMA GEMM: C(f16) = A(f16) @ W^T(f16) + bias [opt relu] ------
// mode 0: dense rows.  mode 1: gather rows via list[t], write compact (off+i).
// mode 2: compact rows (off+i) in and out.  t = blockIdx.z.
// Async double-buffered LDS pipeline; every wave issues exactly 4 async b128 / chunk.
__global__ __launch_bounds__(256)
void gemm128(const _Float16* __restrict__ A, const _Float16* __restrict__ Wt,
             const float* __restrict__ bias, _Float16* __restrict__ C,
             int K, int N, int reluOut, int mode,
             const int* __restrict__ cnt, const int* __restrict__ offs,
             const int* __restrict__ list, int Btot,
             long wTaskStride, int bTaskStride)
{
  const int t = blockIdx.z;
  const _Float16* W = Wt + (size_t)t * (size_t)wTaskStride;
  const float* bv = bias + (size_t)t * (size_t)bTaskStride;
  const int m0 = blockIdx.x * 128;
  const int n0 = blockIdx.y * 128;
  int count = 0x7fffffff, off = 0;
  if (mode != 0) { count = cnt[t]; off = offs[t]; if (m0 >= count) return; }

  __shared__ _Float16 lA[2 * ABUF];
  __shared__ _Float16 lB[2 * BBUF];

  const int tid = threadIdx.x;
  const int lane = tid & 31;
  const int wave = tid >> 5;
  const int wm = wave & 1;       // 2 waves along M (64 rows each)
  const int wn = wave >> 1;      // 4 waves along N (32 cols each)
  const int laneHalf = lane >> 4;
  const int laneN = lane & 15;

  v8f acc[4][2] = {};

  // staging: thread -> (row, 16-half chunk); rows beyond count clamped (discarded at store)
  const int srow = tid >> 1;
  const int sHalf = tid & 1;
  long aRow;
  {
    const int gi = m0 + srow;
    if (mode == 0) aRow = gi;
    else {
      const int cg = gi < count ? gi : count - 1;
      aRow = (mode == 1) ? (long)list[(size_t)t * Btot + cg] : (long)(off + cg);
    }
  }
  const _Float16* aG = A + aRow * (long)K + sHalf * 16;
  const _Float16* bG = W + (size_t)(n0 + srow) * K + sHalf * 16;
  const unsigned aL = lds_addr(lA + srow * LDP + sHalf * 16);
  const unsigned bL = lds_addr(lB + srow * LDP + sHalf * 16);

  const int NK = K >> 5;
#define ISSUE_CHUNK(kc, buf)                                                     \
  do {                                                                           \
    const _Float16* ga = aG + (kc) * 32;                                         \
    const _Float16* gb = bG + (kc) * 32;                                         \
    const unsigned la = aL + (unsigned)(buf) * (ABUF * 2);                       \
    const unsigned lb = bL + (unsigned)(buf) * (BBUF * 2);                       \
    async_b128(la, ga); async_b128(la + 16, ga + 8);                             \
    async_b128(lb, gb); async_b128(lb + 16, gb + 8);                             \
  } while (0)

  ISSUE_CHUNK(0, 0);
  for (int ic = 0; ic < NK; ++ic) {
    const int cur = ic & 1;
    if (ic + 1 < NK) {
      ISSUE_CHUNK(ic + 1, cur ^ 1);
      asm volatile("s_wait_asynccnt 0x4" ::: "memory");  // prev chunk done (in-order)
    } else {
      asm volatile("s_wait_asynccnt 0x0" ::: "memory");
    }
    __syncthreads();

    const _Float16* bufA = lA + cur * ABUF;
    const _Float16* bufB = lB + cur * BBUF;
    HFrag aF[4], bF[2];
    const int kb = laneHalf * 16;
#pragma unroll
    for (int mt = 0; mt < 4; ++mt) {
      const _Float16* pa = bufA + (wm * 64 + mt * 16 + laneN) * LDP;
      aF[mt].q[0] = *(const v4h*)(pa + kb);
      aF[mt].q[1] = *(const v4h*)(pa + kb + 4);
      aF[mt].q[2] = *(const v4h*)(pa + kb + 8);
      aF[mt].q[3] = *(const v4h*)(pa + kb + 12);
    }
#pragma unroll
    for (int nt = 0; nt < 2; ++nt) {
      const _Float16* pb = bufB + (wn * 32 + nt * 16 + laneN) * LDP + laneHalf * 16;
      bF[nt].q[0] = *(const v4h*)(pb + 0);
      bF[nt].q[1] = *(const v4h*)(pb + 4);
      bF[nt].q[2] = *(const v4h*)(pb + 8);
      bF[nt].q[3] = *(const v4h*)(pb + 12);
    }
#pragma unroll
    for (int mt = 0; mt < 4; ++mt)
#pragma unroll
      for (int nt = 0; nt < 2; ++nt)
        acc[mt][nt] = __builtin_amdgcn_wmma_f32_16x16x32_f16(
            false, aF[mt].v, false, bF[nt].v, (short)0, acc[mt][nt], false, false);
    __syncthreads();   // all reads of bufA/bufB done before it is overwritten
  }
#undef ISSUE_CHUNK

  // ---- store C = acc + bias (f16, optional fused relu for the consumer) ----
#pragma unroll
  for (int mt = 0; mt < 4; ++mt) {
#pragma unroll
    for (int nt = 0; nt < 2; ++nt) {
      const int col = n0 + wn * 32 + nt * 16 + laneN;
      const float b = bv[col];
#pragma unroll
      for (int j = 0; j < 8; ++j) {
        const int r = wm * 64 + mt * 16 + laneHalf * 8 + j;
        const int gi = m0 + r;
        float v = acc[mt][nt][j] + b;
        if (reluOut) v = fmaxf(v, 0.0f);
        if (mode == 0) {
          C[(long)gi * N + col] = (_Float16)v;
        } else if (gi < count) {
          C[(long)(off + gi) * N + col] = (_Float16)v;
        }
      }
    }
  }
}

// ---------------- fused head: R(f16, relu'd)@h1W + b1 -> relu -> @h2W + b2 -> sigmoid
__global__ __launch_bounds__(256)
void head_fused(const _Float16* __restrict__ R, const _Float16* __restrict__ H1t,
                const float* __restrict__ h1b, const float* __restrict__ h2w,
                const float* __restrict__ h2b, float* __restrict__ out,
                const int* __restrict__ cnt, const int* __restrict__ offs,
                const int* __restrict__ list, int Btot, int K)
{
  const int t = blockIdx.y;
  const int count = cnt[t];
  const int m0 = blockIdx.x * 128;
  if (m0 >= count) return;
  const int off = offs[t];

  __shared__ _Float16 lA[2 * ABUF];
  __shared__ _Float16 lB[2 * 64 * LDP];
  __shared__ float lH[128 * 68];

  const _Float16* W = H1t + (size_t)t * 64 * K;
  const int tid = threadIdx.x;
  const int lane = tid & 31;
  const int wave = tid >> 5;
  const int wm = wave & 3;       // 4 waves along M (32 rows each)
  const int wn = wave >> 2;      // 2 waves along N (32 cols each)
  const int laneHalf = lane >> 4;
  const int laneN = lane & 15;

  v8f acc[2][2] = {};

  const int srow = tid >> 1;
  const int sHalf = tid & 1;
  const int giS = m0 + srow;
  const int cgS = giS < count ? giS : count - 1;
  const _Float16* aG = R + (size_t)(off + cgS) * K + sHalf * 16;
  const unsigned aL = lds_addr(lA + srow * LDP + sHalf * 16);

  const int brow = tid >> 2;     // 0..63
  const int bq = tid & 3;        // 8 halves (16B) each
  const _Float16* bG = W + (size_t)brow * K + bq * 8;
  const unsigned bL = lds_addr(lB + brow * LDP + bq * 8);

  const int NK = K >> 5;
#define ISSUE_CHUNK_H(kc, buf)                                                   \
  do {                                                                           \
    const _Float16* ga = aG + (kc) * 32;                                         \
    const _Float16* gb = bG + (kc) * 32;                                         \
    const unsigned la = aL + (unsigned)(buf) * (ABUF * 2);                       \
    const unsigned lb = bL + (unsigned)(buf) * (64 * LDP * 2);                   \
    async_b128(la, ga); async_b128(la + 16, ga + 8);                             \
    async_b128(lb, gb);                                                          \
  } while (0)

  ISSUE_CHUNK_H(0, 0);
  for (int ic = 0; ic < NK; ++ic) {
    const int cur = ic & 1;
    if (ic + 1 < NK) {
      ISSUE_CHUNK_H(ic + 1, cur ^ 1);
      asm volatile("s_wait_asynccnt 0x3" ::: "memory");
    } else {
      asm volatile("s_wait_asynccnt 0x0" ::: "memory");
    }
    __syncthreads();

    const _Float16* bufA = lA + cur * ABUF;
    const _Float16* bufB = lB + cur * (64 * LDP);
    HFrag aF[2], bF[2];
    const int kb = laneHalf * 16;
#pragma unroll
    for (int mt = 0; mt < 2; ++mt) {
      const _Float16* pa = bufA + (wm * 32 + mt * 16 + laneN) * LDP;
      aF[mt].q[0] = *(const v4h*)(pa + kb);
      aF[mt].q[1] = *(const v4h*)(pa + kb + 4);
      aF[mt].q[2] = *(const v4h*)(pa + kb + 8);
      aF[mt].q[3] = *(const v4h*)(pa + kb + 12);
    }
#pragma unroll
    for (int nt = 0; nt < 2; ++nt) {
      const _Float16* pb = bufB + (wn * 32 + nt * 16 + laneN) * LDP + laneHalf * 16;
      bF[nt].q[0] = *(const v4h*)(pb + 0);
      bF[nt].q[1] = *(const v4h*)(pb + 4);
      bF[nt].q[2] = *(const v4h*)(pb + 8);
      bF[nt].q[3] = *(const v4h*)(pb + 12);
    }
#pragma unroll
    for (int mt = 0; mt < 2; ++mt)
#pragma unroll
      for (int nt = 0; nt < 2; ++nt)
        acc[mt][nt] = __builtin_amdgcn_wmma_f32_16x16x32_f16(
            false, aF[mt].v, false, bF[nt].v, (short)0, acc[mt][nt], false, false);
    __syncthreads();
  }
#undef ISSUE_CHUNK_H

  // H = relu(acc + b1) into LDS
#pragma unroll
  for (int mt = 0; mt < 2; ++mt)
#pragma unroll
    for (int nt = 0; nt < 2; ++nt) {
      const int c = wn * 32 + nt * 16 + laneN;
      const float b = h1b[t * 64 + c];
#pragma unroll
      for (int j = 0; j < 8; ++j) {
        const int r = wm * 32 + mt * 16 + laneHalf * 8 + j;
        lH[r * 68 + c] = fmaxf(acc[mt][nt][j] + b, 0.0f);
      }
    }
  __syncthreads();

  if (tid < 128) {
    const int g = m0 + tid;
    if (g < count) {
      float s = h2b[t];
      const float* w2 = h2w + t * 64;
#pragma unroll 8
      for (int j = 0; j < 64; ++j) s += lH[tid * 68 + j] * w2[j];
      out[list[(size_t)t * Btot + g]] = 1.0f / (1.0f + expf(-s));
    }
  }
}

// ------------------------------------------------------------------------------------
extern "C" void kernel_launch(void* const* d_in, const int* in_sizes, int n_in,
                              void* d_out, int out_size, void* d_ws, size_t ws_size,
                              hipStream_t stream) {
  const float* x    = (const float*)d_in[0];
  const int*   bt   = (const int*)d_in[1];
  const float* ws1W = (const float*)d_in[2];
  const float* ws1b = (const float*)d_in[3];
  const float* wt1W = (const float*)d_in[4];
  const float* wt1b = (const float*)d_in[5];
  const float* ws2W = (const float*)d_in[6];
  const float* ws2b = (const float*)d_in[7];
  const float* wt2W = (const float*)d_in[8];
  const float* wt2b = (const float*)d_in[9];
  const float* h1W  = (const float*)d_in[10];
  const float* h1b  = (const float*)d_in[11];
  const float* h2W  = (const float*)d_in[12];
  const float* h2b  = (const float*)d_in[13];
  float* out = (float*)d_out;

  const int B = 2048, D = 1024;
  const long MM = (long)D * D;

  // workspace layout (all f16 activations)
  char* ws = (char*)d_ws;
  _Float16* WtSq = (_Float16*)ws;                               // 20 * MM f16 (transposed)
  _Float16* WtH1 = WtSq + (size_t)20 * MM;                      // 4 * 64 * 1024 f16
  _Float16* xh   = WtH1 + (size_t)4 * 64 * 1024;                // [B][D] f16
  _Float16* buf0 = xh   + (size_t)B * D;                        // [B][D] f16
  _Float16* buf1 = buf0 + (size_t)B * D;                        // [B][D] f16
  int* cnt  = (int*)(buf1 + (size_t)B * D);                     // [4]
  int* offs = cnt + 4;                                          // [5]
  int* list = offs + 8;                                         // [4*B]

  dim3 tb(32, 8);
  transpose_sq_f16<<<dim3(32, 32, 20), tb, 0, stream>>>(ws1W, wt1W, ws2W, wt2W, WtSq);
  transpose_h1_f16<<<dim3(2, 32, 4), tb, 0, stream>>>(h1W, WtH1);
  convert_f16<<<dim3((B * D) / (256 * 8)), 256, 0, stream>>>(x, xh);
  build_lists<<<1, 256, 0, stream>>>(bt, B, cnt, offs, list);

  const dim3 gDense(16, 8, 1), gTask(16, 8, 4), blk(256);
  // shared block 1 (fuse consumer relu into producer store)
  gemm128<<<gDense, blk, 0, stream>>>(xh,   WtSq + 0 * MM, ws1b + 0 * D, buf0, D, D, 1, 0, cnt, offs, list, B, 0, 0);
  gemm128<<<gDense, blk, 0, stream>>>(buf0, WtSq + 1 * MM, ws1b + 1 * D, buf1, D, D, 1, 0, cnt, offs, list, B, 0, 0);
  // task block 1 (gather -> compact, then compact); l1 output feeds shared2-l0: NO relu
  gemm128<<<gTask,  blk, 0, stream>>>(buf1, WtSq + 2 * MM, wt1b + 0 * D, buf0, D, D, 1, 1, cnt, offs, list, B, 2 * MM, 2 * D);
  gemm128<<<gTask,  blk, 0, stream>>>(buf0, WtSq + 3 * MM, wt1b + 1 * D, buf1, D, D, 0, 2, cnt, offs, list, B, 2 * MM, 2 * D);
  // shared block 2 (stays in compact row order)
  gemm128<<<gDense, blk, 0, stream>>>(buf1, WtSq + 10 * MM, ws2b + 0 * D, buf0, D, D, 1, 0, cnt, offs, list, B, 0, 0);
  gemm128<<<gDense, blk, 0, stream>>>(buf0, WtSq + 11 * MM, ws2b + 1 * D, buf1, D, D, 1, 0, cnt, offs, list, B, 0, 0);
  // task block 2 body (compact); l1 output feeds head which wants relu(r): fuse it
  gemm128<<<gTask,  blk, 0, stream>>>(buf1, WtSq + 12 * MM, wt2b + 0 * D, buf0, D, D, 1, 2, cnt, offs, list, B, 2 * MM, 2 * D);
  gemm128<<<gTask,  blk, 0, stream>>>(buf0, WtSq + 13 * MM, wt2b + 1 * D, buf1, D, D, 1, 2, cnt, offs, list, B, 2 * MM, 2 * D);
  // fused head + sigmoid + scatter to natural order
  head_fused<<<dim3(16, 4), blk, 0, stream>>>(buf1, WtH1, h1b, h2W, h2b, out, cnt, offs, list, B, D);
}
